// WindowedMultiHeadAttention_32220844654670
// MI455X (gfx1250) — matrix operand order repaired and checked by reading, not compiled, not run on hardware
//
#include <hip/hip_runtime.h>
#include <stdint.h>

// ---------------------------------------------------------------------------
// Windowed multi-head attention for MI455X (gfx1250), bf16 WMMA pipeline.
//   K0: weight transpose/convert (fp32 -> bf16, K-contiguous rows)
//   K1: QKV GEMM  (fp32 x -> bf16 Q/K row-major, V transposed+padded)
//   K2: per-(window,head): S=QK^T -> softmax/sqrt(C) -> O=PV
//       (V^T async-copied global->LDS, overlapped with phases 1-2)
//   K3: proj GEMM + bias + residual + window reverse (fp32 out)
// All matmuls: v_wmma_f32_16x16x32_bf16, fp32 accumulate, b128 operand loads.
// ---------------------------------------------------------------------------

typedef __bf16 bf16;
typedef bf16  bf16x16 __attribute__((ext_vector_type(16)));
typedef float v8f     __attribute__((ext_vector_type(8)));
typedef int   async_v4i __attribute__((vector_size(4 * sizeof(int))));

#define AS1 __attribute__((address_space(1)))
#define AS3 __attribute__((address_space(3)))

#define C_EMB  768
#define NH     8
#define DH     96
#define WSZ    14
#define NTOK   196      // 14*14 tokens per window
#define IMG    56
#define NPOS   3136     // 56*56
#define NB     256      // 16 images * 16 windows
#define MT     13       // ceil(196/16)
#define SS     208      // LDS row stride for S (fp32)
#define PS     224      // padded K (and row stride) for P / V^T, 7*32
#define S_BYTES (NTOK * SS * 4)            // 163,072
#define P_BYTES (NTOK * PS * 2)            //  87,808
#define V_BYTES (DH * PS * 2)              //  43,008
#define LDS_TOTAL (S_BYTES + P_BYTES + V_BYTES)   // 293,888 < 320 KB

#if defined(__has_builtin)
#if __has_builtin(__builtin_amdgcn_global_load_async_to_lds_b128)
#define HAVE_ASYNC_LDS 1
#endif
#endif

union Frag {
    bf16x16      v;
    unsigned int u[8];
    bf16         h[16];
};

// Window-partition addressing: element offset of (win, token r, channel 0)
__device__ __forceinline__ size_t x_index(int win, int r) {
    int b    = win >> 4;                   // 16 windows per image
    int wrem = win & 15;
    int wy   = wrem >> 2, wx = wrem & 3;   // 4x4 windows of 14x14
    int ry   = r / WSZ,  rx = r % WSZ;
    int pos  = (wy * WSZ + ry) * IMG + (wx * WSZ + rx);
    return ((size_t)b * NPOS + pos) * C_EMB;
}

// ---------------------------------------------------------------------------
// K0: weight transpose + fp32->bf16.  wtqkv[j][k] = wqkv[k][j] (j < 2304),
//     wtproj[n][k] = wproj[k][n].  One-time ~9 MB read; rows become
//     K-contiguous so WMMA B fragments are two b128 loads.
// ---------------------------------------------------------------------------
__global__ void wmha_wt_kernel(const float* __restrict__ wqkv,
                               const float* __restrict__ wproj,
                               bf16* __restrict__ wtqkv,
                               bf16* __restrict__ wtproj)
{
    int j = blockIdx.x;
    if (j < 3 * C_EMB) {
        for (int k = threadIdx.x; k < C_EMB; k += blockDim.x)
            wtqkv[(size_t)j * C_EMB + k] = (bf16)wqkv[(size_t)k * (3 * C_EMB) + j];
    } else {
        int n = j - 3 * C_EMB;
        for (int k = threadIdx.x; k < C_EMB; k += blockDim.x)
            wtproj[(size_t)n * C_EMB + k] = (bf16)wproj[(size_t)k * C_EMB + n];
    }
}

// ---------------------------------------------------------------------------
// K1: qkv = window_partition(x) @ w_qkv + b_qkv.
// Q/K scratch: [win][head][196][96] row-major bf16.
// V scratch:   [win][head][96][224] (transposed, tokens 196..223 zeroed).
// One wave per (win, head, comp, mtile): 16x96 output.
// w_qkv column for (head, dd, comp) = head*288 + dd*3 + comp  ("(h d qkv)").
// ---------------------------------------------------------------------------
__global__ void wmha_qkv_kernel(const float* __restrict__ x,
                                const bf16* __restrict__ wtqkv,
                                const float* __restrict__ bqkv,
                                bf16* __restrict__ qs,
                                bf16* __restrict__ ks,
                                bf16* __restrict__ vt)
{
    int wid  = (blockIdx.x * blockDim.x + threadIdx.x) >> 5;
    int lane = threadIdx.x & 31;
    int mtile = wid % MT;
    int t     = wid / MT;
    int comp  = t % 3;  t /= 3;
    int head  = t & 7;
    int win   = t >> 3;
    if (win >= NB) return;

    int lm = lane & 15;
    int hl = lane >> 4;
    int mrow = mtile * 16 + lm;
    if (mrow > NTOK - 1) mrow = NTOK - 1;            // clamp pad rows
    const float* xrow = x + x_index(win, mrow);

    v8f acc[6];
    #pragma unroll
    for (int nt = 0; nt < 6; nt++)
        #pragma unroll
        for (int i = 0; i < 8; i++) acc[nt][i] = 0.0f;

    for (int kb = 0; kb < 24; kb++) {                // K = 768 = 24*32
        int k0 = kb * 32;
        Frag a;
        {   // A 16x32: interleaved K-pair layout, 4x float4 + packed cvt
            float4 f0 = *(const float4*)(xrow + k0 + hl * 8);
            float4 f1 = *(const float4*)(xrow + k0 + hl * 8 + 4);
            float4 f2 = *(const float4*)(xrow + k0 + 16 + hl * 8);
            float4 f3 = *(const float4*)(xrow + k0 + 16 + hl * 8 + 4);
            a.h[0]=(bf16)f0.x;  a.h[1]=(bf16)f0.y;  a.h[2]=(bf16)f0.z;  a.h[3]=(bf16)f0.w;
            a.h[4]=(bf16)f1.x;  a.h[5]=(bf16)f1.y;  a.h[6]=(bf16)f1.z;  a.h[7]=(bf16)f1.w;
            a.h[8]=(bf16)f2.x;  a.h[9]=(bf16)f2.y;  a.h[10]=(bf16)f2.z; a.h[11]=(bf16)f2.w;
            a.h[12]=(bf16)f3.x; a.h[13]=(bf16)f3.y; a.h[14]=(bf16)f3.z; a.h[15]=(bf16)f3.w;
        }
        #pragma unroll
        for (int nt = 0; nt < 6; nt++) {
            int dd = nt * 16 + lm;
            int j  = head * 288 + dd * 3 + comp;     // interleaved qkv column
            const bf16* wrow = wtqkv + (size_t)j * C_EMB;
            Frag b;                                  // B 32x16: K = hl*8 + e, 16 + hl*8 + (e-8)
            *(uint4*)&b.u[0] = *(const uint4*)(wrow + k0 + hl * 8);
            *(uint4*)&b.u[4] = *(const uint4*)(wrow + k0 + 16 + hl * 8);
            acc[nt] = __builtin_amdgcn_wmma_f32_16x16x32_bf16(
                false, a.v, false, b.v, (short)0, acc[nt], false, false);
        }
    }

    if (comp < 2) {   // Q / K: row-major [token][d]
        bf16* dst = (comp == 0) ? qs : ks;
        size_t base = ((size_t)(win * NH + head)) * NTOK * DH;
        #pragma unroll
        for (int nt = 0; nt < 6; nt++) {
            int n = nt * 16 + lm;
            float bias = bqkv[head * 288 + n * 3 + comp];
            #pragma unroll
            for (int i = 0; i < 8; i++) {
                int m = mtile * 16 + hl * 8 + i;     // D: M = i + 8*(lane>=16)
                if (m < NTOK)
                    dst[base + (size_t)m * DH + n] = (bf16)(acc[nt][i] + bias);
            }
        }
    } else {          // V: transposed [d][token], padded to 224 tokens
        size_t base = ((size_t)(win * NH + head)) * DH * PS;
        #pragma unroll
        for (int nt = 0; nt < 6; nt++) {
            int n = nt * 16 + lm;                    // d index
            float bias = bqkv[head * 288 + n * 3 + 2];
            #pragma unroll
            for (int i = 0; i < 8; i++) {
                int m = mtile * 16 + hl * 8 + i;     // token index (<= 207)
                float val = (m < NTOK) ? (acc[nt][i] + bias) : 0.0f;
                vt[base + (size_t)n * PS + m] = (bf16)val;
            }
        }
        if (mtile == MT - 1) {                       // zero tokens 208..223
            for (int idx = lane; idx < DH * 16; idx += 32) {
                int dd = idx >> 4, tok = 208 + (idx & 15);
                vt[base + (size_t)dd * PS + tok] = (bf16)0.0f;
            }
        }
    }
}

// ---------------------------------------------------------------------------
// K2: per-(window, head) attention.
//   - async copy V^T slice (96x224 bf16) global -> LDS at kernel start
//   - S(196x196) = Q K^T   (WMMA, fp32 S in LDS)
//   - P = softmax_row(S) * (1/sqrt(768))   (faithful: softmax THEN scale)
//   - O(196x96) = P V      (WMMA, A from LDS P, B from LDS V^T)
// Dynamic LDS: 293,888 B (< 320 KB/WGP).
// ---------------------------------------------------------------------------
__global__ void wmha_attn_kernel(const bf16* __restrict__ qs,
                                 const bf16* __restrict__ ks,
                                 const bf16* __restrict__ vt,
                                 bf16* __restrict__ os)
{
    extern __shared__ char smem[];
    float* S  = (float*)smem;
    bf16*  P  = (bf16*)(smem + S_BYTES);
    bf16*  Vl = (bf16*)(smem + S_BYTES + P_BYTES);

    int wh   = blockIdx.x;                 // win*8 + head
    int win  = wh >> 3;
    int head = wh & 7;
    size_t baseqk = (size_t)wh * NTOK * DH;
    const bf16* q    = qs + baseqk;
    const bf16* kmat = ks + baseqk;
    const bf16* vsrc = vt + (size_t)wh * DH * PS;

    int wave = threadIdx.x >> 5;
    int lane = threadIdx.x & 31;
    int lm = lane & 15, hl = lane >> 4;

    // ---- Kick off V^T copy to LDS (consumed in phase 3) ----
    {
        const int nchunk = V_BYTES / 16;                 // 2688 x 16B
        for (int c = threadIdx.x; c < nchunk; c += blockDim.x) {
#ifdef HAVE_ASYNC_LDS
            __builtin_amdgcn_global_load_async_to_lds_b128(
                (AS1 async_v4i*)(vsrc + c * 8),
                (AS3 async_v4i*)(Vl + c * 8), 0, 0);
#else
            *(uint4*)(Vl + c * 8) = *(const uint4*)(vsrc + c * 8);
#endif
        }
    }

    // ---- Phase 1: S = Q K^T (169 16x16 tiles over 8 waves) ----
    for (int tile = wave; tile < MT * MT; tile += 8) {
        int mtile = tile / MT, ntile = tile % MT;
        int mrow = mtile * 16 + lm; if (mrow > NTOK - 1) mrow = NTOK - 1;
        int ncol = ntile * 16 + lm; if (ncol > NTOK - 1) ncol = NTOK - 1;
        const bf16* arow = q    + (size_t)mrow * DH;
        const bf16* brow = kmat + (size_t)ncol * DH;
        v8f acc;
        #pragma unroll
        for (int i = 0; i < 8; i++) acc[i] = 0.0f;
        #pragma unroll
        for (int kb = 0; kb < 3; kb++) {             // d = 96 = 3*32
            int k0 = kb * 32;
            Frag a, b;
            *(uint4*)&a.u[0] = *(const uint4*)(arow + k0 + hl * 8);
            *(uint4*)&a.u[4] = *(const uint4*)(arow + k0 + 16 + hl * 8);
            *(uint4*)&b.u[0] = *(const uint4*)(brow + k0 + hl * 8);
            *(uint4*)&b.u[4] = *(const uint4*)(brow + k0 + 16 + hl * 8);
            acc = __builtin_amdgcn_wmma_f32_16x16x32_bf16(
                false, a.v, false, b.v, (short)0, acc, false, false);
        }
        #pragma unroll
        for (int i = 0; i < 8; i++) {
            int m = mtile * 16 + hl * 8 + i;
            int n = ntile * 16 + lm;
            if (m < NTOK && n < NTOK) S[m * SS + n] = acc[i];
        }
    }
    __syncthreads();

    // ---- Phase 2: row softmax, then scale by 1/sqrt(C) ----
    const float inv_sqrt_c = 0.03608439182435161f;   // 1/sqrt(768)
    int row = threadIdx.x;
    if (row < NTOK) {
        float mx = -1e30f;
        for (int j = 0; j < NTOK; j++) mx = fmaxf(mx, S[row * SS + j]);
        float sum = 0.0f;
        for (int j = 0; j < NTOK; j++) {
            float e = __expf(S[row * SS + j] - mx);
            S[row * SS + j] = e;
            sum += e;
        }
        float scale = inv_sqrt_c / sum;
        for (int j = 0; j < NTOK; j++)
            P[row * PS + j] = (bf16)(S[row * SS + j] * scale);
        for (int j = NTOK; j < PS; j++)              // zero-pad K to 224
            P[row * PS + j] = (bf16)0.0f;
    }
#ifdef HAVE_ASYNC_LDS
#if __has_builtin(__builtin_amdgcn_s_wait_asynccnt)
    __builtin_amdgcn_s_wait_asynccnt(0);
#else
    asm volatile("s_wait_asynccnt 0" ::: "memory");
#endif
#endif
    __syncthreads();

    // ---- Phase 3: O = P V (78 16x16 tiles over 8 waves, all-LDS operands) --
    for (int tile = wave; tile < MT * 6; tile += 8) {
        int mtile = tile / 6, ntile = tile % 6;
        int mrow = mtile * 16 + lm; if (mrow > NTOK - 1) mrow = NTOK - 1;
        int n = ntile * 16 + lm;                     // d index
        const bf16* arow = P  + mrow * PS;
        const bf16* brow = Vl + n * PS;              // V^T row: K-contiguous
        v8f acc;
        #pragma unroll
        for (int i = 0; i < 8; i++) acc[i] = 0.0f;
        #pragma unroll
        for (int kb = 0; kb < 7; kb++) {             // K = 224 (padded 196)
            int k0 = kb * 32;
            Frag a, b;
            *(uint4*)&a.u[0] = *(const uint4*)(arow + k0 + hl * 8);
            *(uint4*)&a.u[4] = *(const uint4*)(arow + k0 + 16 + hl * 8);
            *(uint4*)&b.u[0] = *(const uint4*)(brow + k0 + hl * 8);
            *(uint4*)&b.u[4] = *(const uint4*)(brow + k0 + 16 + hl * 8);
            acc = __builtin_amdgcn_wmma_f32_16x16x32_bf16(
                false, a.v, false, b.v, (short)0, acc, false, false);
        }
        #pragma unroll
        for (int i = 0; i < 8; i++) {
            int m = mtile * 16 + hl * 8 + i;
            if (m < NTOK)
                os[((size_t)win * NTOK + m) * C_EMB + head * DH + n] =
                    (bf16)acc[i];
        }
    }
}

// ---------------------------------------------------------------------------
// K3: out = window_reverse(O @ w_proj + b_proj) + x   (fp32 out)
// One wave per (win, mtile, 128-col group): 16x128 output, 8 accumulators.
// ---------------------------------------------------------------------------
__global__ void wmha_proj_kernel(const bf16* __restrict__ os,
                                 const bf16* __restrict__ wtproj,
                                 const float* __restrict__ bproj,
                                 const float* __restrict__ x,
                                 float* __restrict__ out)
{
    int wid  = (blockIdx.x * blockDim.x + threadIdx.x) >> 5;
    int lane = threadIdx.x & 31;
    int ng    = wid % 6;
    int t     = wid / 6;
    int mtile = t % MT;
    int win   = t / MT;
    if (win >= NB) return;

    int lm = lane & 15, hl = lane >> 4;
    int mrow = mtile * 16 + lm; if (mrow > NTOK - 1) mrow = NTOK - 1;
    const bf16* orow = os + ((size_t)win * NTOK + mrow) * C_EMB;
    int nb0 = ng * 128;

    v8f acc[8];
    #pragma unroll
    for (int nt = 0; nt < 8; nt++)
        #pragma unroll
        for (int i = 0; i < 8; i++) acc[nt][i] = 0.0f;

    for (int kb = 0; kb < 24; kb++) {                // K = 768
        int k0 = kb * 32;
        Frag a;
        *(uint4*)&a.u[0] = *(const uint4*)(orow + k0 + hl * 8);
        *(uint4*)&a.u[4] = *(const uint4*)(orow + k0 + 16 + hl * 8);
        #pragma unroll
        for (int nt = 0; nt < 8; nt++) {
            int n = nb0 + nt * 16 + lm;
            const bf16* wrow = wtproj + (size_t)n * C_EMB;
            Frag b;
            *(uint4*)&b.u[0] = *(const uint4*)(wrow + k0 + hl * 8);
            *(uint4*)&b.u[4] = *(const uint4*)(wrow + k0 + 16 + hl * 8);
            acc[nt] = __builtin_amdgcn_wmma_f32_16x16x32_bf16(
                false, a.v, false, b.v, (short)0, acc[nt], false, false);
        }
    }

    #pragma unroll
    for (int nt = 0; nt < 8; nt++) {
        int n = nb0 + nt * 16 + lm;
        float bias = bproj[n];
        #pragma unroll
        for (int i = 0; i < 8; i++) {
            int m = mtile * 16 + hl * 8 + i;
            if (m < NTOK) {
                size_t idx = x_index(win, m) + n;    // window reverse
                out[idx] = acc[nt][i] + bias + x[idx];
            }
        }
    }
}

// ---------------------------------------------------------------------------
// Host launcher. Scratch (bf16): Q | K | V^T | O | wtqkv | wtproj  (~324 MB)
// ---------------------------------------------------------------------------
extern "C" void kernel_launch(void* const* d_in, const int* in_sizes, int n_in,
                              void* d_out, int out_size, void* d_ws, size_t ws_size,
                              hipStream_t stream) {
    const float* x     = (const float*)d_in[0];
    const float* wqkv  = (const float*)d_in[1];
    const float* bqkv  = (const float*)d_in[2];
    const float* wproj = (const float*)d_in[3];
    const float* bproj = (const float*)d_in[4];
    float* out = (float*)d_out;

    const size_t e_qk = (size_t)NB * NH * NTOK * DH;   // 38,535,168
    const size_t e_v  = (size_t)NB * NH * DH * PS;     // 44,040,192
    const size_t e_o  = (size_t)NB * NTOK * C_EMB;     // 38,535,168
    bf16* qs     = (bf16*)d_ws;
    bf16* ks     = qs + e_qk;
    bf16* vt     = ks + e_qk;
    bf16* os     = vt + e_v;
    bf16* wtqkv  = os + e_o;
    bf16* wtproj = wtqkv + (size_t)3 * C_EMB * C_EMB;

    {   // K0: 2304 + 768 rows
        wmha_wt_kernel<<<4 * C_EMB, 256, 0, stream>>>(wqkv, wproj, wtqkv, wtproj);
    }
    {   // K1: 256 win * 8 heads * 3 comps * 13 mtiles = 79,872 waves
        int blocks = (NB * NH * 3 * MT) / 8;
        wmha_qkv_kernel<<<blocks, 256, 0, stream>>>(x, wtqkv, bqkv, qs, ks, vt);
    }
    {   // K2: one workgroup per (win, head)
        wmha_attn_kernel<<<NB * NH, 256, (size_t)LDS_TOTAL, stream>>>(qs, ks, vt, os);
    }
    {   // K3: 256 win * 13 mtiles * 6 col-groups = 19,968 waves
        int blocks = (NB * MT * 6) / 8;
        wmha_proj_kernel<<<blocks, 256, 0, stream>>>(os, wtproj, bproj, x, out);
    }
}
